// TemporalAttention_23441931501726
// MI455X (gfx1250) — matrix operand order repaired and checked by reading, not compiled, not run on hardware
//
#include <hip/hip_runtime.h>
#include <hip/hip_bf16.h>

// ---------------------------------------------------------------------------
// TemporalAttention on MI455X (gfx1250).
// Dominant cost (38.9 GFLOP of ~39.2): qkv = (h@W2+b2) . x rewritten as a
// shared-weight GEMM  qkv[o,tok] = sum_{k=e*64+d} (h[e,tok]*x[d,tok])*W2t[o,k]
// (K=8256, bias folded as an h==1 row). bf16 WMMA, f32 accumulate.
// W2 (6.3MB f32) is pre-transposed to bf16 W2t (3MB, L2-resident). 32-token
// tiles per workgroup halve L2 traffic vs 16 (B fragments reused across two
// token sub-tiles); A fragments are built with v_pk_mul_bf16 (splat(h)*xfrag),
// one LDS scalar + one splat per sub-tile per 12 WMMAs.
// ---------------------------------------------------------------------------

#define D_MODEL   64
#define D_HIDDEN  128
#define NUM_HEADS 8
#define HEAD_A    8
#define BB        4
#define TT        24
#define NN        128
#define NTOK      (BB*TT*NN)                   // 12288
#define OQKV      (3*NUM_HEADS*HEAD_A)         // 192
#define KTOT      (D_HIDDEN*D_MODEL + D_MODEL) // 8256
#define BNHTA     (BB*NN*NUM_HEADS*TT*HEAD_A)  // 786432
#define HTA       (NUM_HEADS*TT*HEAD_A)        // 1536

typedef __attribute__((ext_vector_type(16))) __bf16 v16bf;
typedef __attribute__((ext_vector_type(8)))  float  v8f;

union F16x16 {            // one WMMA 16-bit fragment: 16 bf16 == 8 dwords
  v16bf v;
  unsigned int u[8];
};

__device__ __forceinline__ v8f wmma_bf16(v16bf a, v16bf b, v8f c) {
  return __builtin_amdgcn_wmma_f32_16x16x32_bf16(false, a, false, b,
                                                 (short)0, c, false, false);
}

// af = splat(h) * xf, elementwise bf16 -> 8x v_pk_mul_bf16
__device__ __forceinline__ F16x16 frag_scale(unsigned int hsplat,
                                             const F16x16& xf) {
  F16x16 af;
#pragma unroll
  for (int j = 0; j < 8; ++j) {
    asm("v_pk_mul_bf16 %0, %1, %2" : "=v"(af.u[j]) : "v"(hsplat), "v"(xf.u[j]));
  }
  return af;
}

__device__ __forceinline__ unsigned int bf16_splat2(float f) {
  unsigned short r = __builtin_bit_cast(unsigned short, (__bf16)f);
  return (unsigned int)r * 0x10001u;
}

// ---------------------------------------------------------------------------
// Kernel 0: bf16 transposed weights in workspace.
//   W1t[e][d]      = bf16(W1[d,e])                          (128x64)
//   W2t[o][e*64+d] = bf16(W2[e, o*64+d]);  W2t[o][8192+d] = bf16(b2[o*64+d])
// Rows contiguous in k -> every WMMA B-fragment is one 32B load.
// ---------------------------------------------------------------------------
__global__ __launch_bounds__(256) void prep_kernel(
    const float* __restrict__ W1, const float* __restrict__ W2,
    const float* __restrict__ b2,
    __bf16* __restrict__ W1t, __bf16* __restrict__ W2t) {
  int idx = blockIdx.x * 256 + threadIdx.x;
  const int NW2 = OQKV * KTOT;
  if (idx < NW2) {
    int o = idx / KTOT;
    int k = idx - o * KTOT;
    float v;
    if (k < D_HIDDEN * D_MODEL) {
      int e = k >> 6, d = k & 63;
      v = W2[(size_t)e * (3*D_MODEL*D_MODEL) + o * 64 + d];
    } else {
      int d = k - D_HIDDEN * D_MODEL;
      v = b2[o * 64 + d];
    }
    W2t[idx] = (__bf16)v;
  } else if (idx < NW2 + D_HIDDEN * D_MODEL) {
    int j = idx - NW2;
    int e = j >> 6, d = j & 63;
    W1t[j] = (__bf16)W1[d * D_HIDDEN + e];
  }
}

// ---------------------------------------------------------------------------
// Kernel A: 32-token tile (same b,t; 32 consecutive n), 4 waves.
//   stage1: h = relu(x@W1+b1) via WMMA -> LDS f32 [129][32] (row128 == 1)
//   stage2: loop e=0..128; per e: 2 LDS scalars + 2 splats, per 32-k half:
//           2 pk-mul fragments + 3 B loads -> 6 WMMAs (12 per e).
// ---------------------------------------------------------------------------
__global__ __launch_bounds__(128) void qkv_kernel(
    const float* __restrict__ x, const float* __restrict__ b1,
    const __bf16* __restrict__ W1t, const __bf16* __restrict__ W2t,
    float* __restrict__ qkvws) {
  __shared__ float h_lds[(D_HIDDEN + 1) * 32];

  const int lane = threadIdx.x & 31;
  const int w    = threadIdx.x >> 5;   // wave 0..3
  const int tokl = lane & 15;          // A-matrix M lane
  const int g    = lane >> 4;          // K half group
  const int tile0 = blockIdx.x * 32;

  // --- x fragments (A-layout 16x32, bf16) for both token sub-tiles ---
  F16x16 xf[2][2];                     // [subtile][k-half]
#pragma unroll
  for (int s = 0; s < 2; ++s) {
    const float* xr = x + (size_t)(tile0 + s * 16 + tokl) * D_MODEL;
#pragma unroll
    for (int v = 0; v < 8; ++v) {
      int k0 = (v << 1) + (g << 4);
      float2 lo = *reinterpret_cast<const float2*>(xr + k0);
      float2 hi = *reinterpret_cast<const float2*>(xr + 32 + k0);
      xf[s][0].v[2*v]   = (__bf16)lo.x;  xf[s][0].v[2*v+1] = (__bf16)lo.y;
      xf[s][1].v[2*v]   = (__bf16)hi.x;  xf[s][1].v[2*v+1] = (__bf16)hi.y;
    }
  }

  // --- stage 1: h (wave w owns e-tiles 2w, 2w+1; B frags shared by subtiles)
#pragma unroll
  for (int s = 0; s < 2; ++s) {
    int e = (w * 2 + s) * 16 + tokl;           // B-matrix column (lane)
    const __bf16* wr = W1t + e * 64 + g * 16;
    v16bf bA = *reinterpret_cast<const v16bf*>(wr);       // d 0..31
    v16bf bB = *reinterpret_cast<const v16bf*>(wr + 32);  // d 32..63
    v8f a0 = {}, a1 = {};
    a0 = wmma_bf16(xf[0][0].v, bA, a0);
    a0 = wmma_bf16(xf[0][1].v, bB, a0);
    a1 = wmma_bf16(xf[1][0].v, bA, a1);
    a1 = wmma_bf16(xf[1][1].v, bB, a1);
    float bias = b1[e];
#pragma unroll
    for (int r = 0; r < 8; ++r) {              // C: token = r + 8*g
      float h0 = a0[r] + bias, h1 = a1[r] + bias;
      h_lds[e * 32 + (r + 8*g)]      = h0 > 0.f ? h0 : 0.f;
      h_lds[e * 32 + 16 + (r + 8*g)] = h1 > 0.f ? h1 : 0.f;
    }
  }
  if (threadIdx.x < 32) h_lds[D_HIDDEN * 32 + threadIdx.x] = 1.0f; // bias row
  __syncthreads();

  // --- stage 2: hot GEMM. Wave owns o-tiles 3w..3w+2, both sub-tiles. ---
  v8f acc[2][3] = {};
  const int o0 = w * 3 * 16;
  size_t ob[3];
#pragma unroll
  for (int j = 0; j < 3; ++j)
    ob[j] = (size_t)(o0 + j * 16 + tokl) * KTOT + g * 16;

  for (int e = 0; e <= D_HIDDEN; ++e) {
    unsigned int hs0 = bf16_splat2(h_lds[e * 32 + tokl]);
    unsigned int hs1 = bf16_splat2(h_lds[e * 32 + 16 + tokl]);
    const int kbase = e * 64;
#pragma unroll
    for (int half = 0; half < 2; ++half) {
      F16x16 af0 = frag_scale(hs0, xf[0][half]);
      F16x16 af1 = frag_scale(hs1, xf[1][half]);
      const int koff = kbase + half * 32;
#pragma unroll
      for (int j = 0; j < 3; ++j) {
        v16bf bf = *reinterpret_cast<const v16bf*>(W2t + ob[j] + koff);
        acc[0][j] = wmma_bf16(af0.v, bf, acc[0][j]);
        acc[1][j] = wmma_bf16(af1.v, bf, acc[1][j]);
      }
    }
    if (e < 120) __builtin_prefetch(W2t + ob[0] + kbase + 512, 0, 1);
  }

  // --- scatter qkv -> [c3][b][n][hh][t][a] workspace ---
  const int tb = tile0 / (TT * NN);
  const int tt = (tile0 / NN) % TT;
  const int n0 = tile0 % NN;
#pragma unroll
  for (int s = 0; s < 2; ++s) {
#pragma unroll
    for (int j = 0; j < 3; ++j) {
      int o  = o0 + j * 16 + tokl;
      int c3 = o >> 6, hh = (o >> 3) & 7, a = o & 7;
      float* dst = qkvws + (size_t)c3 * BNHTA + (size_t)tb * NN * HTA
                 + hh * (TT * HEAD_A) + tt * HEAD_A + a;
#pragma unroll
      for (int r = 0; r < 8; ++r) {
        int n = n0 + s * 16 + r + 8 * g;       // C: token row = r + 8*g
        dst[(size_t)n * HTA] = acc[s][j][r];
      }
    }
  }
}

// ---------------------------------------------------------------------------
// Kernel B: attention over T=24 per (b,n,head); 8 heads per 256-thread block,
// lane t owns one query row. ~38 MFLOP total -> fp32 VALU.
// ---------------------------------------------------------------------------
__global__ __launch_bounds__(256) void attn_kernel(
    const float* __restrict__ qkvws, float* __restrict__ attnws) {
  int id = blockIdx.x * 8 + (threadIdx.x >> 5);   // (b*N + n)*H + hh
  int hh = id & 7, n = (id >> 3) & 127, b = id >> 10;
  int t = threadIdx.x & 31;
  const float* Q = qkvws + (size_t)id * (TT * HEAD_A);
  const float* K = Q + BNHTA;
  const float* V = K + BNHTA;
  if (t < TT) {
    float4 q0 = *reinterpret_cast<const float4*>(Q + t * 8);
    float4 q1 = *reinterpret_cast<const float4*>(Q + t * 8 + 4);
    float sc[TT];
    float m = -1e30f;
#pragma unroll
    for (int s = 0; s < TT; ++s) {
      float4 k0 = *reinterpret_cast<const float4*>(K + s * 8);
      float4 k1 = *reinterpret_cast<const float4*>(K + s * 8 + 4);
      float d = q0.x*k0.x + q0.y*k0.y + q0.z*k0.z + q0.w*k0.w
              + q1.x*k1.x + q1.y*k1.y + q1.z*k1.z + q1.w*k1.w;
      d *= 0.35355339059327373f;            // 1/sqrt(8)
      sc[s] = d;
      m = fmaxf(m, d);
    }
    float sum = 0.f;
#pragma unroll
    for (int s = 0; s < TT; ++s) { sc[s] = __expf(sc[s] - m); sum += sc[s]; }
    float inv = 1.f / sum;
    float4 o0 = {0,0,0,0}, o1 = {0,0,0,0};
#pragma unroll
    for (int s = 0; s < TT; ++s) {
      float p = sc[s] * inv;
      float4 v0 = *reinterpret_cast<const float4*>(V + s * 8);
      float4 v1 = *reinterpret_cast<const float4*>(V + s * 8 + 4);
      o0.x += p * v0.x; o0.y += p * v0.y; o0.z += p * v0.z; o0.w += p * v0.w;
      o1.x += p * v1.x; o1.y += p * v1.y; o1.z += p * v1.z; o1.w += p * v1.w;
    }
    float* dst = attnws + ((((size_t)b * TT + t) * NN + n) * D_MODEL + hh * HEAD_A);
    *reinterpret_cast<float4*>(dst)     = o0;
    *reinterpret_cast<float4*>(dst + 4) = o1;
  }
}

// ---------------------------------------------------------------------------
// Kernel C: out = LN(attn@Wo + x). One wave per token, lane owns cols
// {lane, lane+32}; wave32 shuffle reductions (warpSize==32 on CDNA5).
// ---------------------------------------------------------------------------
__global__ __launch_bounds__(256) void proj_ln_kernel(
    const float* __restrict__ attnws, const float* __restrict__ x,
    const float* __restrict__ Wo, const float* __restrict__ ln_w,
    const float* __restrict__ ln_b, float* __restrict__ out) {
  int w = threadIdx.x >> 5, lane = threadIdx.x & 31;
  int tg = blockIdx.x * 8 + w;
  const float* ar = attnws + (size_t)tg * D_MODEL;
  float acc0 = 0.f, acc1 = 0.f;
#pragma unroll 4
  for (int d4 = 0; d4 < 16; ++d4) {
    float4 av = *reinterpret_cast<const float4*>(ar + d4 * 4);
    float ads[4] = {av.x, av.y, av.z, av.w};
#pragma unroll
    for (int q = 0; q < 4; ++q) {
      int d = d4 * 4 + q;
      acc0 += ads[q] * Wo[d * D_MODEL + lane];
      acc1 += ads[q] * Wo[d * D_MODEL + lane + 32];
    }
  }
  float y0 = acc0 + x[(size_t)tg * D_MODEL + lane];
  float y1 = acc1 + x[(size_t)tg * D_MODEL + lane + 32];
  float s = y0 + y1;
#pragma unroll
  for (int off = 16; off > 0; off >>= 1) s += __shfl_xor(s, off, 32);
  float mu = s * (1.0f / 64.0f);
  float d0 = y0 - mu, d1 = y1 - mu;
  float v = d0 * d0 + d1 * d1;
#pragma unroll
  for (int off = 16; off > 0; off >>= 1) v += __shfl_xor(v, off, 32);
  float rstd = rsqrtf(v * (1.0f / 64.0f) + 1e-5f);
  out[(size_t)tg * D_MODEL + lane]      = d0 * rstd * ln_w[lane]      + ln_b[lane];
  out[(size_t)tg * D_MODEL + lane + 32] = d1 * rstd * ln_w[lane + 32] + ln_b[lane + 32];
}

// ---------------------------------------------------------------------------
extern "C" void kernel_launch(void* const* d_in, const int* in_sizes, int n_in,
                              void* d_out, int out_size, void* d_ws, size_t ws_size,
                              hipStream_t stream) {
  const float* x   = (const float*)d_in[0];
  const float* W1  = (const float*)d_in[1];
  const float* b1  = (const float*)d_in[2];
  const float* W2  = (const float*)d_in[3];
  const float* b2  = (const float*)d_in[4];
  const float* Wo  = (const float*)d_in[5];
  const float* lnw = (const float*)d_in[6];
  const float* lnb = (const float*)d_in[7];

  char* ws = (char*)d_ws;
  // workspace layout (256B aligned):
  __bf16* W1t    = (__bf16*)(ws + 0);          // 128*64*2   = 16384 B
  __bf16* W2t    = (__bf16*)(ws + 16384);      // 192*8256*2 = 3170304 B
  float*  qkvws  = (float*)(ws + 3186688);     // 3*786432*4 = 9437184 B
  float*  attnws = (float*)(ws + 12623872);    // 786432*4   = 3145728 B

  const int prepN = OQKV * KTOT + D_HIDDEN * D_MODEL;
  prep_kernel<<<(prepN + 255) / 256, 256, 0, stream>>>(W1, W2, b2, W1t, W2t);
  qkv_kernel<<<NTOK / 32, 128, 0, stream>>>(x, b1, W1t, W2t, qkvws);
  attn_kernel<<<BB * NN * NUM_HEADS / 8, 256, 0, stream>>>(qkvws, attnws);
  proj_ln_kernel<<<NTOK / 8, 256, 0, stream>>>(attnws, x, Wo, lnw, lnb,
                                               (float*)d_out);
}
